// SelfAttention_39676907882805
// MI455X (gfx1250) — matrix operand order repaired and checked by reading, not compiled, not run on hardware
//
#include <hip/hip_runtime.h>

typedef __attribute__((ext_vector_type(16))) __bf16 v16bf;
typedef __attribute__((ext_vector_type(8)))  __bf16 v8bf;
typedef __attribute__((ext_vector_type(8)))  float  v8f;

// Convert 8 contiguous f32 at p into bf16 slots [base, base+7] of dst.
__device__ __forceinline__ void load_cvt8(const float* __restrict__ p, v16bf& dst, int base) {
    const float4 a = *reinterpret_cast<const float4*>(p);
    const float4 b = *reinterpret_cast<const float4*>(p + 4);
    dst[base + 0] = (__bf16)a.x; dst[base + 1] = (__bf16)a.y;
    dst[base + 2] = (__bf16)a.z; dst[base + 3] = (__bf16)a.w;
    dst[base + 4] = (__bf16)b.x; dst[base + 5] = (__bf16)b.y;
    dst[base + 6] = (__bf16)b.z; dst[base + 7] = (__bf16)b.w;
}

// C[M,N] = A[M,K] @ W[N,K]^T + bias, converting f32 operands to bf16 fragments
// on the fly. Each wave computes a 32x64 tile: 2 M-tiles x 4 N-tiles, so every
// B fragment feeds two WMMAs and every A fragment feeds four (L2 intensity
// ~10.7 FLOP/B vs 6.4 for a 16x64 tile). Block = 256 threads = 8 waves
// covering 32 (M) x 512 (N). Grid = (M/32, N/512).
template <bool OUT_BF16>
__global__ __launch_bounds__(256)
void gemm_xwt_bias(const float* __restrict__ A, const float* __restrict__ W,
                   const float* __restrict__ bias, void* __restrict__ Cout,
                   int M, int N, int K) {
    const int lane = threadIdx.x & 31;
    const int wave = threadIdx.x >> 5;
    const int half = lane >> 4;
    const int l16  = lane & 15;
    const int m0 = blockIdx.x << 5;
    const int n0 = (blockIdx.y << 9) + (wave << 6);

    // A-fragment rows for this lane (lanes 0-15 and 16-31 both map M=0..15;
    // they differ only in which K-slots they hold: +0/+16 vs +8/+24).
    const float* arow0 = A + (size_t)(m0 + l16) * K + (half << 3);
    const float* arow1 = arow0 + (size_t)16 * K;

    v8f z = {0.f, 0.f, 0.f, 0.f, 0.f, 0.f, 0.f, 0.f};
    v8f acc[2][4] = {{z, z, z, z}, {z, z, z, z}};

    for (int kk = 0; kk < K; kk += 32) {
        v16bf af0, af1;
        load_cvt8(arow0 + kk,      af0, 0);   // K = kk + 8*half + 0..7
        load_cvt8(arow0 + kk + 16, af0, 8);   // K = kk + 16 + 8*half + 0..7
        load_cvt8(arow1 + kk,      af1, 0);
        load_cvt8(arow1 + kk + 16, af1, 8);
#pragma unroll
        for (int t = 0; t < 4; ++t) {
            // B[k,n] = W[n,k]: per lane K = kk + 8*half + 0..7 and kk + 16 + 8*half + 0..7.
            const float* brow = W + (size_t)(n0 + (t << 4) + l16) * K + kk + (half << 3);
            v16bf bfv;
            load_cvt8(brow,     bfv, 0);
            load_cvt8(brow + 16, bfv, 8);
            acc[0][t] = __builtin_amdgcn_wmma_f32_16x16x32_bf16(
                false, af0, false, bfv, (short)0, acc[0][t], false, false);
            acc[1][t] = __builtin_amdgcn_wmma_f32_16x16x32_bf16(
                false, af1, false, bfv, (short)0, acc[1][t], false, false);
        }
    }

#pragma unroll
    for (int t = 0; t < 4; ++t) {
        const int n  = n0 + (t << 4) + l16;
        const float bv = bias[n];
#pragma unroll
        for (int mt = 0; mt < 2; ++mt) {
#pragma unroll
            for (int r = 0; r < 8; ++r) {
                const int mr = m0 + (mt << 4) + (half << 3) + r;  // C/D: M = r (+8 hi half)
                const float v = acc[mt][t][r] + bv;
                if (OUT_BF16)
                    ((__bf16*)Cout)[(size_t)mr * N + n] = (__bf16)v;
                else
                    ((float*)Cout)[(size_t)mr * N + n] = v;
            }
        }
    }
}

// Fused per-(b,h) attention: scores = (Q Kt^T)/64, softmax, dropout, AV.
// One wave per block; grid = B*H = 2048. S=32, D=128.
__global__ __launch_bounds__(32)
void attn_kernel(const __bf16* __restrict__ Q, const __bf16* __restrict__ Kt,
                 const __bf16* __restrict__ V, const float* __restrict__ mask,
                 float* __restrict__ O) {
    __shared__ __attribute__((aligned(32))) float  sS[32 * 33];   // padded: conflict-free rows
    __shared__ __attribute__((aligned(32))) __bf16 sA[32 * 32];   // softmaxed attn (bf16)
    __shared__ __attribute__((aligned(32))) __bf16 sVT[128 * 32]; // V^T: [d][s]

    const int lane = threadIdx.x;
    const int half = lane >> 4;
    const int l16  = lane & 15;
    const int b = blockIdx.x >> 5;   // H == 32
    const int h = blockIdx.x & 31;
    const size_t headoff = (size_t)(b * 32) * 4096 + (size_t)h * 128;

    // Stage V^T into LDS (B-operand of AV is N-contiguous in memory otherwise).
    {
        const __bf16* vrow = V + headoff + (size_t)lane * 4096;  // row s = lane
#pragma unroll 8
        for (int d = 0; d < 128; ++d) sVT[d * 32 + lane] = vrow[d];
    }

    // ---- scores: 2x2 tiles of 16x16, K = D = 128 in 4 steps of 32 ----
    v8f z = {0.f, 0.f, 0.f, 0.f, 0.f, 0.f, 0.f, 0.f};
    v8f acc[2][2] = {{z, z}, {z, z}};
    for (int kk = 0; kk < 128; kk += 32) {
        v16bf af[2];
#pragma unroll
        for (int mt = 0; mt < 2; ++mt) {
            const __bf16* qr = Q + headoff + (size_t)(mt * 16 + l16) * 4096 + kk + half * 16;
            v8bf lo = *reinterpret_cast<const v8bf*>(qr);
            v8bf hi = *reinterpret_cast<const v8bf*>(qr + 8);
            af[mt] = __builtin_shufflevector(lo, hi, 0, 1, 2, 3, 4, 5, 6, 7,
                                             8, 9, 10, 11, 12, 13, 14, 15);
        }
#pragma unroll
        for (int nt = 0; nt < 2; ++nt) {
            // B[k,n] = K_head[n,k]: 16 contiguous bf16 along k per lane.
            const __bf16* kr = Kt + headoff + (size_t)(nt * 16 + l16) * 4096 + kk + half * 16;
            v16bf bfv = *reinterpret_cast<const v16bf*>(kr);
#pragma unroll
            for (int mt = 0; mt < 2; ++mt)
                acc[mt][nt] = __builtin_amdgcn_wmma_f32_16x16x32_bf16(
                    false, af[mt], false, bfv, (short)0, acc[mt][nt], false, false);
        }
    }
    // Scale by 1/sqrt(E)=1/64 and stash. (-100000 row constant cancels in softmax.)
#pragma unroll
    for (int mt = 0; mt < 2; ++mt)
#pragma unroll
        for (int nt = 0; nt < 2; ++nt)
#pragma unroll
            for (int r = 0; r < 8; ++r)
                sS[(mt * 16 + half * 8 + r) * 33 + nt * 16 + l16] =
                    acc[mt][nt][r] * 0.015625f;
    __syncthreads();

    // ---- softmax + dropout: thread `lane` owns row s_q = lane ----
    {
        const float* mrow = mask + ((size_t)(b * 32 + h) * 32 + lane) * 32;
        float mx = -3.4e38f;
        for (int j = 0; j < 32; ++j) mx = fmaxf(mx, sS[lane * 33 + j]);
        float ev[32];
        float sum = 0.f;
        for (int j = 0; j < 32; ++j) {
            const float e = __expf(sS[lane * 33 + j] - mx);
            ev[j] = e; sum += e;
        }
        const float inv = 1.f / sum;
        for (int j = 0; j < 32; ++j)
            sA[lane * 32 + j] = (__bf16)(ev[j] * inv * mrow[j]);
    }
    __syncthreads();

    // ---- AV: out[32x128] = attn[32x32] @ V[32x128], K = 32 (one WMMA step) ----
    v16bf af[2];
#pragma unroll
    for (int mt = 0; mt < 2; ++mt) {
        const __bf16* ar = sA + (mt * 16 + l16) * 32 + half * 16;
        v8bf lo = *reinterpret_cast<const v8bf*>(ar);
        v8bf hi = *reinterpret_cast<const v8bf*>(ar + 8);
        af[mt] = __builtin_shufflevector(lo, hi, 0, 1, 2, 3, 4, 5, 6, 7,
                                         8, 9, 10, 11, 12, 13, 14, 15);
    }
#pragma unroll
    for (int nt = 0; nt < 8; ++nt) {
        const __bf16* br = sVT + (nt * 16 + l16) * 32 + half * 16;  // B[k,n] = vT[n][k]
        v16bf bfv = *reinterpret_cast<const v16bf*>(br);
        v8f o0 = z, o1 = z;
        o0 = __builtin_amdgcn_wmma_f32_16x16x32_bf16(false, af[0], false, bfv, (short)0, o0, false, false);
        o1 = __builtin_amdgcn_wmma_f32_16x16x32_bf16(false, af[1], false, bfv, (short)0, o1, false, false);
#pragma unroll
        for (int r = 0; r < 8; ++r) {
            const int n = nt * 16 + l16;
            O[headoff + (size_t)(half * 8 + r) * 4096 + n]      = o0[r];
            O[headoff + (size_t)(16 + half * 8 + r) * 4096 + n] = o1[r];
        }
    }
}

extern "C" void kernel_launch(void* const* d_in, const int* in_sizes, int n_in,
                              void* d_out, int out_size, void* d_ws, size_t ws_size,
                              hipStream_t stream) {
    (void)in_sizes; (void)n_in; (void)out_size; (void)ws_size;
    const float* x1    = (const float*)d_in[0];
    const float* x2    = (const float*)d_in[1];
    const float* wq    = (const float*)d_in[2];
    const float* bq    = (const float*)d_in[3];
    const float* wk    = (const float*)d_in[4];
    const float* bk    = (const float*)d_in[5];
    const float* wv    = (const float*)d_in[6];
    const float* bv    = (const float*)d_in[7];
    const float* wo    = (const float*)d_in[8];
    const float* bo    = (const float*)d_in[9];
    const float* dmask = (const float*)d_in[10];

    char* ws = (char*)d_ws;
    const size_t MS = 2048ull * 4096ull;  // tokens x E elements
    __bf16* qb = (__bf16*)(ws);            // bf16: 16 MB
    __bf16* kb = (__bf16*)(ws + MS * 2);   // bf16: 16 MB
    __bf16* vb = (__bf16*)(ws + MS * 4);   // bf16: 16 MB
    float*  ao = (float*)(ws + MS * 6);    // f32 : 33 MB  (total ~84 MB)

    dim3 gblk(256), ggrid(64, 8);          // M/32 x N/512
    // k from x1; q, v from x2 (as in the source)
    gemm_xwt_bias<true><<<ggrid, gblk, 0, stream>>>(x1, wk, bk, (void*)kb, 2048, 4096, 4096);
    gemm_xwt_bias<true><<<ggrid, gblk, 0, stream>>>(x2, wq, bq, (void*)qb, 2048, 4096, 4096);
    gemm_xwt_bias<true><<<ggrid, gblk, 0, stream>>>(x2, wv, bv, (void*)vb, 2048, 4096, 4096);

    attn_kernel<<<dim3(2048), dim3(32), 0, stream>>>(qb, kb, vb, dmask, ao);

    gemm_xwt_bias<false><<<ggrid, gblk, 0, stream>>>(ao, wo, bo, d_out, 2048, 4096, 4096);
}
